// SepCNNBlock_17085379904261
// MI455X (gfx1250) — matrix operand order repaired and checked by reading, not compiled, not run on hardware
//
#include <hip/hip_runtime.h>

// CDNA5 / gfx1250: wave32, WMMA 16x16x32 bf16 -> f32 accumulate.
// Batched GEMM: out[b] = W(512x512) * x[b](512x2048) + bias, B=32.
// f32 inputs are decomposed on the fly into bf16 hi/lo; 3 WMMAs per tile
// (hi*hi + hi*lo + lo*hi) give ~2^-17 relative error (f32-faithful).

typedef __attribute__((ext_vector_type(16))) __bf16 v16bf;
typedef __attribute__((ext_vector_type(4)))  __bf16 v4bf;
typedef __attribute__((ext_vector_type(8)))  float  v8f;

#define B_DIM 32
#define H_DIM 512
#define E_DIM 512
#define T_DIM 2048

#define BM 128   // E tile per block
#define BN 256   // T tile per block
#define BK 32    // K step (one bf16 WMMA K-depth)

// Fragment-ordered LDS tiles: [16x16 tile index][lane 0..31][elem 0..15]
// A (16x32 bf16, contiguous k): lane = (m&15) + 16*(k>>4), elem = (k&15)
// B (32x16 bf16, contiguous k): lane = (n&15) + 16*(k>>4),     elem = (k&15)

__global__ __launch_bounds__(256)
void conv1x1_gemm_bf16x2_wmma(const float* __restrict__ x,
                              const float* __restrict__ W,
                              const float* __restrict__ bias,
                              float* __restrict__ out)
{
    __shared__ __align__(32) __bf16 Ahi[8][32][16];    //  8 KB
    __shared__ __align__(32) __bf16 Alo[8][32][16];    //  8 KB
    __shared__ __align__(32) __bf16 Bhi[16][32][16];   // 16 KB
    __shared__ __align__(32) __bf16 Blo[16][32][16];   // 16 KB

    const int tid    = threadIdx.x;
    const int lane   = tid & 31;
    const int wave   = tid >> 5;     // 0..7
    const int wave_m = wave >> 2;    // 0..1 -> 64 rows of E each
    const int wave_n = wave & 3;     // 0..3 -> 64 cols of T each

    const int b  = blockIdx.z;
    const int e0 = blockIdx.y * BM;
    const int t0 = blockIdx.x * BN;

    const float* __restrict__ xb = x + (size_t)b * H_DIM * T_DIM;

    // A-tile loader: 2 threads per row, 4 float4 each (128 rows x 32 k)
    const int a_row   = tid >> 1;            // 0..127
    const int a_kbase = (tid & 1) * 16;      // 0 or 16
    const int a_tile  = a_row >> 4;
    const int a_m15   = a_row & 15;

    // B-tile loader: thread owns one column n = tid, all 32 k (coalesced b32)
    const int b_n    = tid;                  // 0..255
    const int b_tile = b_n >> 4;             // 0..15
    const int b_n15  = b_n & 15;

    const v8f vzero = {0.f,0.f,0.f,0.f,0.f,0.f,0.f,0.f};
    v8f acc[4][4];
#pragma unroll
    for (int i = 0; i < 4; ++i)
#pragma unroll
        for (int j = 0; j < 4; ++j) acc[i][j] = vzero;

    for (int k0 = 0; k0 < H_DIM; k0 += BK) {
        // ---- Stage W tile (128x32 f32) -> split bf16, fragment order ----
        {
            const float* __restrict__ wrow =
                W + (size_t)(e0 + a_row) * H_DIM + k0 + a_kbase;
#pragma unroll
            for (int q = 0; q < 4; ++q) {
                const int k  = a_kbase + 4 * q;           // quad base k
                const float4 w4 = *(const float4*)(wrow + 4 * q);
                v4bf h, l;
                h[0] = (__bf16)w4.x; l[0] = (__bf16)(w4.x - (float)h[0]);
                h[1] = (__bf16)w4.y; l[1] = (__bf16)(w4.y - (float)h[1]);
                h[2] = (__bf16)w4.z; l[2] = (__bf16)(w4.z - (float)h[2]);
                h[3] = (__bf16)w4.w; l[3] = (__bf16)(w4.w - (float)h[3]);
                const int wl = a_m15 + ((k >> 4) << 4);        // fragment lane
                const int el = k & 15;                         // fragment elem base
                *(v4bf*)&Ahi[a_tile][wl][el] = h;              // ds_store_b64
                *(v4bf*)&Alo[a_tile][wl][el] = l;
            }
        }
        // ---- Stage X tile (32x256 f32) -> split bf16, fragment order ----
        {
            const float* __restrict__ xcol = xb + (size_t)k0 * T_DIM + t0 + b_n;
#pragma unroll
            for (int q = 0; q < 8; ++q) {                 // k quad = 4q..4q+3
                v4bf h, l;
#pragma unroll
                for (int j = 0; j < 4; ++j) {
                    const float f = xcol[(size_t)(4 * q + j) * T_DIM]; // coalesced b32
                    h[j] = (__bf16)f;
                    l[j] = (__bf16)(f - (float)h[j]);
                }
                const int wl = b_n15 + ((q >> 2) << 4);   // khalf = q>>2
                const int el = (4 * q) & 15;              // k&15, 4 consecutive
                *(v4bf*)&Bhi[b_tile][wl][el] = h;         // ds_store_b64
                *(v4bf*)&Blo[b_tile][wl][el] = l;
            }
        }
        __syncthreads();

        // ---- Compute: 16 tiles x 3 WMMA (hi*hi + hi*lo + lo*hi) ----
        v16bf a_h[4], a_l[4];
#pragma unroll
        for (int mt = 0; mt < 4; ++mt) {
            const int at = wave_m * 4 + mt;
            a_h[mt] = *(const v16bf*)&Ahi[at][lane][0];   // 2x ds_load_b128
            a_l[mt] = *(const v16bf*)&Alo[at][lane][0];
        }
#pragma unroll
        for (int nt = 0; nt < 4; ++nt) {
            const int bt = wave_n * 4 + nt;
            const v16bf b_h = *(const v16bf*)&Bhi[bt][lane][0];
            const v16bf b_l = *(const v16bf*)&Blo[bt][lane][0];
#pragma unroll
            for (int mt = 0; mt < 4; ++mt) {
                acc[mt][nt] = __builtin_amdgcn_wmma_f32_16x16x32_bf16(
                    false, a_h[mt], false, b_h, (short)0, acc[mt][nt], false, false);
                acc[mt][nt] = __builtin_amdgcn_wmma_f32_16x16x32_bf16(
                    false, a_h[mt], false, b_l, (short)0, acc[mt][nt], false, false);
                acc[mt][nt] = __builtin_amdgcn_wmma_f32_16x16x32_bf16(
                    false, a_l[mt], false, b_h, (short)0, acc[mt][nt], false, false);
            }
        }
        __syncthreads();
    }

    // ---- Epilogue: bias add + store (C/D layout: VGPR r -> M = r + 8*(lane>=16)) ----
    const int n_idx = lane & 15;
    const int m_off = (lane >> 4) * 8;
    float* __restrict__ outb = out + (size_t)b * E_DIM * T_DIM;
#pragma unroll
    for (int mt = 0; mt < 4; ++mt) {
        const int e_base = e0 + wave_m * 64 + mt * 16 + m_off;
#pragma unroll
        for (int nt = 0; nt < 4; ++nt) {
            const int tcol = t0 + wave_n * 64 + nt * 16 + n_idx;
            float* __restrict__ op = outb + tcol;
#pragma unroll
            for (int r = 0; r < 8; ++r) {
                const int e = e_base + r;
                op[(size_t)e * T_DIM] = acc[mt][nt][r] + bias[e];
            }
        }
    }
}

extern "C" void kernel_launch(void* const* d_in, const int* in_sizes, int n_in,
                              void* d_out, int out_size, void* d_ws, size_t ws_size,
                              hipStream_t stream) {
    const float* x    = (const float*)d_in[0];
    const float* W    = (const float*)d_in[1];
    const float* bias = (const float*)d_in[2];
    float* out        = (float*)d_out;

    dim3 grid(T_DIM / BN, E_DIM / BM, B_DIM);   // (8, 4, 32)
    dim3 block(256);
    conv1x1_gemm_bf16x2_wmma<<<grid, block, 0, stream>>>(x, W, bias, out);
}
